// MaskRCNN4D_87617332838953
// MI455X (gfx1250) — matrix operand order repaired and checked by reading, not compiled, not run on hardware
//
#include <hip/hip_runtime.h>
#include <hip/hip_bf16.h>
#include <math.h>

// ---------------- problem constants ----------------
#define T_    8
#define D_    8
#define H_    32
#define W_    32
#define NSP   65536          // T*D*H*W
#define C_    32
#define A_    9
#define TOPK  500
#define NSCORE (A_*NSP)      // 589824
#define NMS_THR 0.7f

// ---------------- vector types ----------------
typedef __attribute__((ext_vector_type(16))) __bf16          v16bf;
typedef __attribute__((ext_vector_type(16))) unsigned short  v16s;
typedef __attribute__((ext_vector_type(8)))  unsigned short  v8s;
typedef __attribute__((ext_vector_type(8)))  float           v8f;
typedef __attribute__((ext_vector_type(4)))  unsigned int    v4u;
typedef __attribute__((ext_vector_type(8)))  int             v8i;
typedef __attribute__((ext_vector_type(4)))  int             v4i;

#define HAVE_TDM  __has_builtin(__builtin_amdgcn_tensor_load_to_lds)
#define HAVE_TWAIT __has_builtin(__builtin_amdgcn_s_wait_tensorcnt)

// ---------------- helpers ----------------
static __device__ __forceinline__ unsigned short f2bf(float f) {
  unsigned int u = __float_as_uint(f);
  unsigned int r = u + 0x7FFFu + ((u >> 16) & 1u);   // round-to-nearest-even
  return (unsigned short)(r >> 16);
}
static __device__ __forceinline__ float bf2f(unsigned short h) {
  return __uint_as_float(((unsigned int)h) << 16);
}
// K index for 16-bit A-matrix 16x32 fragment (contiguous order k = 16*(lane/16) + h), lane l, half h(0..15)
static __device__ __forceinline__ int a_frag_k(int lane, int h) {
  return ((lane < 16) ? 0 : 16) + h;
}

// TDM: async DMA of one contiguous 1KB bf16 chunk (512 elem) global -> LDS.
// 1-D descriptor: data_size=2B, tile_dim0=tensor_dim0=512, type=2 ("image").
static __device__ __forceinline__ void tdm_issue_1kb(
    const unsigned short* gsrc, unsigned short* lbuf, int lane) {
#if HAVE_TDM
  unsigned long long ga = (unsigned long long)(const void*)gsrc;
  unsigned lds_addr = (unsigned)(unsigned long long)(void*)lbuf;
  v4u g0 = { 1u,                                   // count=1 (valid user D#)
             lds_addr,                              // lds_addr [63:32]
             (unsigned)ga,                          // global_addr[31:0]
             ((unsigned)(ga >> 32) & 0x01FFFFFFu)   // global_addr[56:32]
               | (2u << 30) };                      // type=2
  v8i g1 = { 0x10000,        // wg_mask=0, data_size=1 (2 bytes)
             512 << 16,      // tensor_dim0[15:0] << 16  (=512)
             0x00010000,     // tensor_dim0 hi=0 | tensor_dim1=1
             512 << 16,      // tensor_dim1 hi=0 | tile_dim0=512
             1,              // tile_dim1=1, tile_dim2=0
             512,            // tensor_dim0_stride low32
             0, 0 };
  v4i gz = { 0, 0, 0, 0 };
#if __clang_major__ >= 23
  v8i gz8 = { 0, 0, 0, 0, 0, 0, 0, 0 };
  __builtin_amdgcn_tensor_load_to_lds(g0, g1, gz, gz, gz8, 0);
#else
  __builtin_amdgcn_tensor_load_to_lds(g0, g1, gz, gz, 0);
#endif
#else
  // fallback: synchronous vector copy (32B per lane x 32 lanes = 1KB)
  *reinterpret_cast<v16s*>(lbuf + lane * 16) =
      *reinterpret_cast<const v16s*>(gsrc + lane * 16);
#endif
}
#if HAVE_TDM && HAVE_TWAIT
#define TDM_WAIT(n) __builtin_amdgcn_s_wait_tensorcnt(n)
#else
#define TDM_WAIT(n) do {} while (0)
#endif

// ---------------- ws layout (bytes) ----------------
#define OFF_WFRAG   0u          // 81*2*32*16 ushort = 165888 B
#define OFF_CLSFRAG 165888u     // 32*16 ushort      = 1024 B
#define OFF_HID     166912u     // 65536*32 ushort   = 4194304 B
#define OFF_SCORES  4361216u    // 589824 float      = 2359296 B
#define OFF_TV      6720512u    // 500 float
#define OFF_TI      6722512u    // 500 int
#define OFF_PROPS   6724512u    // 4000 float
#define OFF_IOU     6740512u    // 250000 float

// ==================================================================
// Kernel 1: pack conv weights + cls weights into WMMA A-fragments
// ==================================================================
__global__ __launch_bounds__(256) void prep_weights(
    const float* __restrict__ conv_w,     // [32][32][81]
    const float* __restrict__ cls_w,      // [9][32]
    unsigned short* __restrict__ wfrag,   // [81][2][32][16]
    unsigned short* __restrict__ clsfrag) // [32][16]
{
  int i = blockIdx.x * 256 + threadIdx.x;
  if (i < 81 * 2 * 32 * 16) {
    int h    = i & 15;
    int lane = (i >> 4) & 31;
    int mtk  = i >> 9;
    int mt   = mtk & 1;
    int k    = mtk >> 1;
    int m    = lane & 15;
    int K    = a_frag_k(lane, h);
    int o    = mt * 16 + m;
    wfrag[i] = f2bf(conv_w[(o * 32 + K) * 81 + k]);
  } else if (i < 81 * 2 * 32 * 16 + 512) {
    int ii   = i - 81 * 2 * 32 * 16;
    int h    = ii & 15;
    int lane = ii >> 4;
    int m    = lane & 15;
    int K    = a_frag_k(lane, h);
    float v  = (m < A_) ? cls_w[m * 32 + K] : 0.0f;
    clsfrag[ii] = f2bf(v);
  }
}

// ==================================================================
// Kernel 2: 4D conv 3^4 via implicit GEMM on v_wmma_f32_16x16x32_bf16.
// One block per output row (t,d,h) = 32 positions. 2 waves, one per
// oc-tile; each wave owns both 16-position sub-tiles (dual accumulators,
// A-fragment reused by 2 WMMAs). Weight stream (1KB/tap/wave) is
// double-buffered into LDS by the Tensor Data Mover, synced with
// s_wait_tensorcnt.
// ==================================================================
__global__ __launch_bounds__(64) void conv4d_wmma(
    const float* __restrict__ feat,
    const float* __restrict__ conv_b,
    const unsigned short* __restrict__ wfrag,
    unsigned short* __restrict__ hid)   // [NSP][32] bf16
{
  __shared__ __align__(32) unsigned short slab[27 * 34 * 32]; // [sp27][w34][cin32]
  __shared__ __align__(32) unsigned short wbuf[2][2][512];    // [wave][pingpong][1KB]
  const int tid = threadIdx.x;
  const int row = blockIdx.x;          // (t*8+d)*32+h
  const int h0  = row & 31;
  const int d0  = (row >> 5) & 7;
  const int t0  = row >> 8;

  // stage padded input slab, transpose to cin-innermost bf16
  for (int i = tid; i < 27 * 34 * 32; i += 64) {
    int c   = i / (27 * 34);
    int rem = i - c * (27 * 34);
    int sp  = rem / 34;
    int wi  = rem - sp * 34;
    int kt = sp / 9, kd = (sp % 9) / 3, kh = sp % 3;
    int tt = t0 + kt - 1, dd = d0 + kd - 1, hh = h0 + kh - 1, ww = wi - 1;
    float v = 0.0f;
    if (tt >= 0 && tt < T_ && dd >= 0 && dd < D_ &&
        hh >= 0 && hh < H_ && ww >= 0 && ww < W_)
      v = feat[((c * T_ + tt) * D_ + dd) * (H_ * W_) + hh * W_ + ww];
    slab[(sp * 34 + wi) * 32 + c] = f2bf(v);
  }
  __syncthreads();

  const int mt    = tid >> 5;          // wave id == oc tile
  const int lane  = tid & 31;
  const int col   = lane & 15;
  const int kbase = (lane < 16) ? 0 : 16;

  // kick off tap 0 weight DMA
  tdm_issue_1kb(wfrag + (size_t)(0 * 2 + mt) * 512, &wbuf[mt][0][0], lane);

  v8f acc0 = {0.f, 0.f, 0.f, 0.f, 0.f, 0.f, 0.f, 0.f};
  v8f acc1 = {0.f, 0.f, 0.f, 0.f, 0.f, 0.f, 0.f, 0.f};

  for (int k = 0; k < 81; ++k) {
    if (k < 80) {
      tdm_issue_1kb(wfrag + (size_t)((k + 1) * 2 + mt) * 512,
                    &wbuf[mt][(k + 1) & 1][0], lane);
      TDM_WAIT(1);      // double-buffer: oldest (tap k) complete
    } else {
      TDM_WAIT(0);      // last tap: drain
    }
    v16s as = *reinterpret_cast<const v16s*>(&wbuf[mt][k & 1][lane * 16]);
    v16bf a = __builtin_bit_cast(v16bf, as);
    int kdiv3 = k / 3, kw = k - kdiv3 * 3;
    const unsigned short* bp = &slab[((kdiv3 * 34) + col + kw) * 32 + kbase];
    v16s b0s = *reinterpret_cast<const v16s*>(bp);
    v16s b1s = *reinterpret_cast<const v16s*>(bp + 16 * 32);
    v16bf b0 = __builtin_bit_cast(v16bf, b0s);
    v16bf b1 = __builtin_bit_cast(v16bf, b1s);
    acc0 = __builtin_amdgcn_wmma_f32_16x16x32_bf16(
        false, a, false, b0, (short)0, acc0, false, false);
    acc1 = __builtin_amdgcn_wmma_f32_16x16x32_bf16(
        false, a, false, b1, (short)0, acc1, false, false);
  }

  // epilogue: bias + ReLU, bf16 store (C/D layout: M = mbase+r, N = col)
  const int mbase = (lane < 16) ? 0 : 8;
  const int pos0  = row * 32 + col;
  v8s ov0, ov1;
#pragma unroll
  for (int r = 0; r < 8; ++r) {
    int oc   = mt * 16 + mbase + r;
    float bv = conv_b[oc];
    float v0 = acc0[r] + bv;
    float v1 = acc1[r] + bv;
    ov0[r] = f2bf(v0 > 0.0f ? v0 : 0.0f);
    ov1[r] = f2bf(v1 > 0.0f ? v1 : 0.0f);
  }
  *reinterpret_cast<v8s*>(hid + (size_t)pos0 * 32 + mt * 16 + mbase) = ov0;
  *reinterpret_cast<v8s*>(hid + (size_t)(pos0 + 16) * 32 + mt * 16 + mbase) = ov1;
}

// ==================================================================
// Kernel 3: cls 1x1 conv (one WMMA per 16 positions) + sigmoid scores
// ==================================================================
__global__ __launch_bounds__(256) void cls_scores_wmma(
    const unsigned short* __restrict__ hid,
    const unsigned short* __restrict__ clsfrag,
    const float* __restrict__ cls_b,
    float* __restrict__ scores)   // [9][NSP], destructively consumed by topk
{
  const int tid   = threadIdx.x;
  const int lane  = tid & 31;
  const int tile  = blockIdx.x * 8 + (tid >> 5);
  const int col   = lane & 15;
  const int kbase = (lane < 16) ? 0 : 16;
  const int pos   = tile * 16 + col;

  v16s as = *reinterpret_cast<const v16s*>(clsfrag + lane * 16);
  v16s bs = *reinterpret_cast<const v16s*>(hid + (size_t)pos * 32 + kbase);
  v16bf a = __builtin_bit_cast(v16bf, as);
  v16bf b = __builtin_bit_cast(v16bf, bs);
  v8f acc = {0.f, 0.f, 0.f, 0.f, 0.f, 0.f, 0.f, 0.f};
  acc = __builtin_amdgcn_wmma_f32_16x16x32_bf16(
      false, a, false, b, (short)0, acc, false, false);

  const int mbase = (lane < 16) ? 0 : 8;
#pragma unroll
  for (int r = 0; r < 8; ++r) {
    int ach = mbase + r;
    if (ach < A_) {
      float lg = acc[r] + cls_b[ach];
      scores[(size_t)ach * NSP + pos] = 1.0f / (1.0f + expf(-lg));
    }
  }
}

// ==================================================================
// Kernel 4: top-500 by iterative block-wide argmax (destructive mark)
// ==================================================================
__global__ __launch_bounds__(1024) void topk_kernel(
    float* __restrict__ scores, float* __restrict__ tv, int* __restrict__ ti)
{
  __shared__ float rv[1024];
  __shared__ int   ri[1024];
  const int tid = threadIdx.x;
  for (int r = 0; r < TOPK; ++r) {
    float best = -INFINITY; int bi = 0;
    for (int i = tid; i < NSCORE; i += 1024) {
      float s = scores[i];
      if (s > best) { best = s; bi = i; }      // strict > => lowest index wins
    }
    rv[tid] = best; ri[tid] = bi;
    __syncthreads();
    for (int s = 512; s > 0; s >>= 1) {
      if (tid < s) {
        if (rv[tid + s] > rv[tid] ||
            (rv[tid + s] == rv[tid] && ri[tid + s] < ri[tid])) {
          rv[tid] = rv[tid + s]; ri[tid] = ri[tid + s];
        }
      }
      __syncthreads();
    }
    if (tid == 0) {
      tv[r] = rv[0]; ti[r] = ri[0];
      scores[ri[0]] = -INFINITY;
    }
    __threadfence_block();
    __syncthreads();
  }
}

// ==================================================================
// Kernel 5: anchors + quirky delta gather (deltas recomputed on the fly)
// ==================================================================
__global__ __launch_bounds__(512) void gather_props(
    const float* __restrict__ tv, const int* __restrict__ ti,
    const unsigned short* __restrict__ hid,
    const float* __restrict__ box_w, const float* __restrict__ box_b,
    float* __restrict__ out, float* __restrict__ props)
{
  const int r = threadIdx.x;
  if (r >= TOPK) return;
  const float sz[3]  = {8.0f, 16.0f, 32.0f};
  const float r2t[3] = {1.0f, 0.5f, 2.0f};
  const float r3t[3] = {1.0f, 2.0f, 0.5f};

  int idx = ti[r];
  int a = idx >> 16;
  int n = idx & 65535;
  int w = n & 31, h = (n >> 5) & 31, d = (n >> 10) & 7, t = n >> 13;
  float ct = t * 4.0f + 2.0f, cd = d * 4.0f + 2.0f;
  float ch = h * 4.0f + 2.0f, cw = w * 4.0f + 2.0f;
  int si = a / 3, ri_ = a - si * 3;
  float s  = sz[si] * 0.5f;
  float h2 = s * r2t[ri_], h3 = s * r3t[ri_];
  float anc[8] = {ct - s, cd - s, ch - h2, cw - h3,
                  ct + s, cd + s, ch + h2, cw + h3};
#pragma unroll
  for (int j = 0; j < 8; ++j) {
    int f  = idx * 8 + j;              // faithful .view(B,-1,8) indexing
    int bc = f >> 16;                  // box channel 0..71
    int nn = f & 65535;                // spatial cell
    float acc = box_b[bc];
    const unsigned short* hc = hid + (size_t)nn * 32;
    for (int c = 0; c < 32; ++c) acc += box_w[bc * 32 + c] * bf2f(hc[c]);
    float p = anc[j] + acc;
    out[r * 8 + j]   = p;
    props[r * 8 + j] = p;
  }
  out[TOPK * 8 + r] = tv[r];           // scores
}

// ==================================================================
// Kernel 6: 500x500 pairwise 4D IoU (volumes unclamped, like reference)
// ==================================================================
__global__ __launch_bounds__(512) void iou_kernel(
    const float* __restrict__ props, float* __restrict__ iou)
{
  const int i = blockIdx.x;
  const int j = threadIdx.x;
  if (j >= TOPK) return;
  const float* A = props + i * 8;
  const float* B = props + j * 8;
  float inter = 1.0f; bool ok = true;
  float va = 1.0f, vb = 1.0f;
#pragma unroll
  for (int dd = 0; dd < 4; ++dd) {
    float lo = fmaxf(A[dd], B[dd]);
    float hi = fminf(A[4 + dd], B[4 + dd]);
    float df = hi - lo;
    ok &= (df > 0.0f);
    inter *= df;
    va *= (A[4 + dd] - A[dd]);
    vb *= (B[4 + dd] - B[dd]);
  }
  inter = ok ? inter : 0.0f;
  iou[i * TOPK + j] = inter / (va + vb - inter + 1e-6f);
}

// ==================================================================
// Kernel 7: sequential greedy NMS
// ==================================================================
__global__ __launch_bounds__(512) void nms_kernel(
    const float* __restrict__ iou, float* __restrict__ out)
{
  __shared__ int keep[TOPK];
  const int tid = threadIdx.x;
  if (tid < TOPK) keep[tid] = 1;
  __syncthreads();
  for (int i = 0; i < TOPK; ++i) {
    int ki = keep[i];
    if (ki && tid < TOPK && tid > i && iou[i * TOPK + tid] > NMS_THR)
      keep[tid] = 0;
    __syncthreads();
  }
  if (tid < TOPK) out[TOPK * 8 + TOPK + tid] = keep[tid] ? 1.0f : 0.0f;
}

// ==================================================================
extern "C" void kernel_launch(void* const* d_in, const int* in_sizes, int n_in,
                              void* d_out, int out_size, void* d_ws, size_t ws_size,
                              hipStream_t stream) {
  (void)in_sizes; (void)n_in; (void)out_size; (void)ws_size;
  const float* feat   = (const float*)d_in[0];
  const float* conv_w = (const float*)d_in[1];
  const float* conv_b = (const float*)d_in[2];
  const float* cls_w  = (const float*)d_in[3];
  const float* cls_b  = (const float*)d_in[4];
  const float* box_w  = (const float*)d_in[5];
  const float* box_b  = (const float*)d_in[6];
  float* out = (float*)d_out;

  char* ws = (char*)d_ws;
  unsigned short* wfrag   = (unsigned short*)(ws + OFF_WFRAG);
  unsigned short* clsfrag = (unsigned short*)(ws + OFF_CLSFRAG);
  unsigned short* hid     = (unsigned short*)(ws + OFF_HID);
  float* scores = (float*)(ws + OFF_SCORES);
  float* tv     = (float*)(ws + OFF_TV);
  int*   ti     = (int*)  (ws + OFF_TI);
  float* props  = (float*)(ws + OFF_PROPS);
  float* iou    = (float*)(ws + OFF_IOU);

  prep_weights<<<327, 256, 0, stream>>>(conv_w, cls_w, wfrag, clsfrag);
  conv4d_wmma<<<T_ * D_ * H_, 64, 0, stream>>>(feat, conv_b, wfrag, hid);
  cls_scores_wmma<<<512, 256, 0, stream>>>(hid, clsfrag, cls_b, scores);
  topk_kernel<<<1, 1024, 0, stream>>>(scores, tv, ti);
  gather_props<<<1, 512, 0, stream>>>(tv, ti, hid, box_w, box_b, out, props);
  iou_kernel<<<TOPK, 512, 0, stream>>>(props, iou);
  nms_kernel<<<1, 512, 0, stream>>>(iou, out);
}
